// TSK_46050639348192
// MI455X (gfx1250) — matrix operand order repaired and checked by reading, not compiled, not run on hardware
//
#include <hip/hip_runtime.h>

typedef __attribute__((ext_vector_type(16))) _Float16     v16h;
typedef __attribute__((ext_vector_type(8)))  float        v8f;
typedef __attribute__((ext_vector_type(4)))  unsigned int v4u;
typedef __attribute__((ext_vector_type(8)))  int          v8i;
typedef __attribute__((ext_vector_type(4)))  int          v4i;

#define HCONST 0.5f
#define EPSV   1e-8f

constexpr int Bn  = 16384;
constexpr int IN  = 64;
constexpr int NR  = 128;
constexpr int OUT = 32;
constexpr int KW  = (IN + 1) * NR; // 8320

// ---------------------------------------------------------------------------
// Kernel 1: fold centers/sigmas into S1 = s, S2 = 2*c*s, Kc[r] = sum_i c^2 s,
// and convert W to f16 (L2-resident, reused 1024x by the WMMA kernel).
// ---------------------------------------------------------------------------
__global__ void prep_kernel(const float* __restrict__ centers,
                            const float* __restrict__ sigmas,
                            const float* __restrict__ W,
                            float* __restrict__ S1, float* __restrict__ S2,
                            float* __restrict__ Kc, _Float16* __restrict__ Wh) {
  int tid    = blockIdx.x * blockDim.x + threadIdx.x;
  int stride = gridDim.x * blockDim.x;
  for (int idx = tid; idx < IN * NR; idx += stride) {
    float sg = sigmas[idx];
    float s  = HCONST / (sg * sg) + EPSV;
    S1[idx] = s;
    S2[idx] = 2.0f * centers[idx] * s;
  }
  for (int r = tid; r < NR; r += stride) {
    float acc = 0.0f;
    for (int i = 0; i < IN; ++i) {
      float c  = centers[i * NR + r];
      float sg = sigmas[i * NR + r];
      float s  = HCONST / (sg * sg) + EPSV;
      acc += c * c * s;
    }
    Kc[r] = acc;
  }
  for (int idx = tid; idx < OUT * KW; idx += stride) {
    Wh[idx] = (_Float16)W[idx];
  }
}

// ---------------------------------------------------------------------------
// Kernel 2: logits (f32, precision-sensitive) + softmax -> frs [B,128].
// logit[b,r] = sum_i x*(S2 - S1*x) - Kc[r].
// ---------------------------------------------------------------------------
__global__ void __launch_bounds__(128)
frs_kernel(const float* __restrict__ X, const float* __restrict__ S1,
           const float* __restrict__ S2, const float* __restrict__ Kc,
           float* __restrict__ frs) {
  __shared__ float xs[16][IN];
  __shared__ float lg[16][NR];
  __shared__ float pm[16][8], rmax[16], ps[16][8], rsum[16];

  const int t    = threadIdx.x;        // rule index
  const int row0 = blockIdx.x * 16;

  for (int e = t; e < 16 * IN; e += 128)
    xs[e / IN][e % IN] = X[(size_t)(row0 + e / IN) * IN + (e % IN)];
  __syncthreads();

  float acc[16];
  const float kc = Kc[t];
#pragma unroll
  for (int m = 0; m < 16; ++m) acc[m] = -kc;
  for (int i = 0; i < IN; ++i) {
    float s1 = S1[i * NR + t], s2 = S2[i * NR + t];
#pragma unroll
    for (int m = 0; m < 16; ++m) {
      float x = xs[m][i];
      acc[m] += x * (s2 - s1 * x);
    }
  }
#pragma unroll
  for (int m = 0; m < 16; ++m) lg[m][t] = acc[m];
  __syncthreads();

  const int m = t >> 3, j = t & 7;
  float mx = -1e30f;
  for (int r = j; r < NR; r += 8) mx = fmaxf(mx, lg[m][r]);
  pm[m][j] = mx;
  __syncthreads();
  if (j == 0) {
    float v = pm[m][0];
#pragma unroll
    for (int q = 1; q < 8; ++q) v = fmaxf(v, pm[m][q]);
    rmax[m] = v;
  }
  __syncthreads();
  float sm = 0.0f;
  const float rm = rmax[m];
  for (int r = j; r < NR; r += 8) sm += expf(lg[m][r] - rm);
  ps[m][j] = sm;
  __syncthreads();
  if (j == 0) {
    float v = 0.0f;
#pragma unroll
    for (int q = 0; q < 8; ++q) v += ps[m][q];
    rsum[m] = v;
  }
  __syncthreads();
  for (int mm = 0; mm < 16; ++mm) {
    float inv = 1.0f / rsum[mm];
    frs[(size_t)(row0 + mm) * NR + t] = expf(lg[mm][t] - rmax[mm]) * inv;
  }
}

// ---------------------------------------------------------------------------
// Kernel 3: fused out = [frs (x) X, frs] @ Wh^T + b via v_wmma_f32_16x16x32_f16.
// W staged by the Tensor Data Mover, double-buffered: wave 0 prefetches stage
// s+1 into buf[(s+1)&1], then s_wait_tensorcnt(1) guarantees stage s landed
// while the next tile flies during compute. Stage s<32: 2-D tile 32 x 256
// halves (4 rules), stride 8320. Stage 32: order-0 tile 32 x 128 halves.
// ---------------------------------------------------------------------------
__global__ void __launch_bounds__(128)
tsk_main_kernel(const float* __restrict__ X, const float* __restrict__ frs,
                const _Float16* __restrict__ Wh, const float* __restrict__ bias,
                float* __restrict__ out) {
  __shared__ _Float16 fls[4][16][132];              // frs tiles in f16 (A precision)
  __shared__ __align__(32) _Float16 wst[2][8192];   // 2 x 16 KB TDM landing buffers

  const int t    = threadIdx.x;
  const int w    = t >> 5;
  const int lane = t & 31;
  const int M    = lane & 15;    // A row selector
  const int kh   = lane >> 4;    // K-half selector
  const int col  = lane & 15;    // B/D column selector
  const int rowbase = blockIdx.x * 64 + w * 16;

  // stage this wave's frs tile [16][128] into LDS as f16
  for (int e = lane; e < 16 * NR; e += 32) {
    int m = e >> 7, r = e & 127;
    fls[w][m][r] = (_Float16)frs[(size_t)(rowbase + m) * NR + r];
  }

  // load + convert this lane's X fragment (contiguous k order: element e holds k = 16*kh + e)
  v16h xA, xB;
  {
    const float* xrow = X + (size_t)(rowbase + M) * IN;
    float tmp[32];
#pragma unroll
    for (int run = 0; run < 4; ++run) {
      const float4* p = (const float4*)(xrow + (run >> 1) * 32 + kh * 16 + (run & 1) * 8);
      float4 a = p[0], b2 = p[1];
      tmp[run * 8 + 0] = a.x;  tmp[run * 8 + 1] = a.y;
      tmp[run * 8 + 2] = a.z;  tmp[run * 8 + 3] = a.w;
      tmp[run * 8 + 4] = b2.x; tmp[run * 8 + 5] = b2.y;
      tmp[run * 8 + 6] = b2.z; tmp[run * 8 + 7] = b2.w;
    }
#pragma unroll
    for (int e = 0; e < 16; ++e) {
      xA[e] = (_Float16)tmp[e];
      xB[e] = (_Float16)tmp[16 + e];
    }
  }

  v8f c0 = {0.f, 0.f, 0.f, 0.f, 0.f, 0.f, 0.f, 0.f};
  v8f c1 = c0;

  // --- TDM descriptor skeleton (D# per CDNA5 ISA) ---
  const unsigned int  lds_base = (unsigned int)(uintptr_t)(void*)wst;
  const unsigned long long gbase = (unsigned long long)(uintptr_t)Wh;
  v4i gz4 = {0, 0, 0, 0};
  v8i gz8 = {0, 0, 0, 0, 0, 0, 0, 0};
  v8i g1;                                  // group 1: dims/strides
  g1[0] = (int)(1u << 16);                 // data_size=1 (2 bytes), wg_mask=0
  g1[1] = (int)((unsigned)(KW & 0xFFFF) << 16);      // tensor_dim0[15:0]=8320
  g1[2] = (int)(32u << 16);                // tensor_dim0 hi=0 | tensor_dim1=32
  g1[3] = (int)(256u << 16);               // tensor_dim1 hi=0 | tile_dim0=256
  g1[4] = 32;                              // tile_dim1=32, tile_dim2=0
  g1[5] = KW;                              // tensor_dim0_stride = 8320
  g1[6] = 0;
  g1[7] = 0;

  auto issue = [&](int s) {                // stage s in [0,32]
    unsigned long long ga = (s < 32)
        ? gbase + (unsigned long long)s * 512u          // 4 rules * 64 * 2B
        : gbase + (unsigned long long)(IN * NR) * 2u;   // order-0 block
    v8i g1l = g1;
    if (s == 32) g1l[3] = (int)(128u << 16);            // tile_dim0 = 128
    v4u g0;
    g0[0] = 1u;                                         // count=1, user D#
    g0[1] = lds_base + (unsigned)((s & 1) * 16384);     // LDS dest buffer
    g0[2] = (unsigned int)ga;                           // global_addr[31:0]
    g0[3] = (unsigned int)((ga >> 32) & 0x1FFFFFFu) | (2u << 30); // hi | type=2
    __builtin_amdgcn_tensor_load_to_lds(g0, g1l, gz4, gz4, gz8, 0);
  };

  if (w == 0) issue(0);

  // ---- rule-major part: 32 pipeline stages of 4 rules each ----
  for (int s = 0; s < 32; ++s) {
    if (w == 0) {
      issue(s + 1);                            // prefetch next stage (s=31 -> order-0)
      __builtin_amdgcn_s_wait_tensorcnt(1);    // stage s landed; s+1 in flight
    }
    __syncthreads();
    const _Float16* wb = wst[s & 1];
#pragma unroll
    for (int rl = 0; rl < 4; ++rl) {
      const int r = s * 4 + rl;
      const _Float16 fh = fls[w][M][r];
      v16h a0, a1;
#pragma unroll
      for (int e = 0; e < 16; ++e) { a0[e] = xA[e] * fh; a1[e] = xB[e] * fh; }
      // B tiles in TDM layout wb[o][rl*64 + i]: 32B contiguous per lane
      v16h b00 = *(const v16h*)(wb + (size_t)col * 256 + rl * 64 + kh * 16);
      v16h b10 = *(const v16h*)(wb + (size_t)col * 256 + rl * 64 + 32 + kh * 16);
      v16h b01 = *(const v16h*)(wb + (size_t)(16 + col) * 256 + rl * 64 + kh * 16);
      v16h b11 = *(const v16h*)(wb + (size_t)(16 + col) * 256 + rl * 64 + 32 + kh * 16);
      c0 = __builtin_amdgcn_wmma_f32_16x16x32_f16(false, a0, false, b00, (short)0, c0, false, false);
      c1 = __builtin_amdgcn_wmma_f32_16x16x32_f16(false, a0, false, b01, (short)0, c1, false, false);
      c0 = __builtin_amdgcn_wmma_f32_16x16x32_f16(false, a1, false, b10, (short)0, c0, false, false);
      c1 = __builtin_amdgcn_wmma_f32_16x16x32_f16(false, a1, false, b11, (short)0, c1, false, false);
    }
    __syncthreads();                           // no wave still reads buf being refilled
  }

  // ---- order-0 part: stage 32 (buffer 0), K = 128 firing levels ----
  if (w == 0) __builtin_amdgcn_s_wait_tensorcnt(0);
  __syncthreads();
  {
    const _Float16* wb = wst[0];
#pragma unroll
    for (int cidx = 0; cidx < 4; ++cidx) {
      v16h a;
#pragma unroll
      for (int e = 0; e < 8; ++e) {
        a[e]     = fls[w][M][cidx * 32 + kh * 16 + e];
        a[8 + e] = fls[w][M][cidx * 32 + kh * 16 + 8 + e];
      }
      v16h b0 = *(const v16h*)(wb + (size_t)col * NR + cidx * 32 + kh * 16);
      v16h b1 = *(const v16h*)(wb + (size_t)(16 + col) * NR + cidx * 32 + kh * 16);
      c0 = __builtin_amdgcn_wmma_f32_16x16x32_f16(false, a, false, b0, (short)0, c0, false, false);
      c1 = __builtin_amdgcn_wmma_f32_16x16x32_f16(false, a, false, b1, (short)0, c1, false, false);
    }
  }

  // ---- epilogue: bias + store (D layout: VGPR v -> row v+8*kh, col lane%16)
  const float bz0 = bias[col], bz1 = bias[16 + col];
#pragma unroll
  for (int v = 0; v < 8; ++v) {
    int row = rowbase + v + 8 * kh;
    out[(size_t)row * OUT + col]      = c0[v] + bz0;
    out[(size_t)row * OUT + 16 + col] = c1[v] + bz1;
  }
}

// ---------------------------------------------------------------------------
extern "C" void kernel_launch(void* const* d_in, const int* in_sizes, int n_in,
                              void* d_out, int out_size, void* d_ws, size_t ws_size,
                              hipStream_t stream) {
  const float* X       = (const float*)d_in[0];
  const float* centers = (const float*)d_in[1];
  const float* sigmas  = (const float*)d_in[2];
  const float* W       = (const float*)d_in[3];
  const float* bias    = (const float*)d_in[4];

  float* out = (float*)d_out;                  // [16384, 32]
  float* frs = out + (size_t)Bn * OUT;         // [16384, 128] (second output)

  char* ws = (char*)d_ws;
  float*    S1 = (float*)(ws);                 //  32 KB
  float*    S2 = (float*)(ws + 32768);         //  32 KB
  float*    Kc = (float*)(ws + 65536);         //  512 B
  _Float16* Wh = (_Float16*)(ws + 66048);      //  520 KB (f16 W, L2 resident)

  prep_kernel<<<520, 256, 0, stream>>>(centers, sigmas, W, S1, S2, Kc, Wh);
  frs_kernel<<<Bn / 16, 128, 0, stream>>>(X, S1, S2, Kc, frs);
  tsk_main_kernel<<<Bn / 64, 128, 0, stream>>>(X, frs, Wh, bias, out);
}
